// _GNN_64613488001134
// MI455X (gfx1250) — matrix operand order repaired and checked by reading, not compiled, not run on hardware
//
#include <hip/hip_runtime.h>
#include <hip/hip_bf16.h>
#include <math.h>

typedef __attribute__((ext_vector_type(16))) _Float16 v16h;
typedef __attribute__((ext_vector_type(8)))  float    v8f;

#define NN      1024
#define D_IN    36
#define KNN     8
#define D_EDGE  32
#define NODE_ST 112      // padded node row stride (cols 0..35 x, 36..67 e1, 68..99 e2)
#define M_PAIRS 523776   // 1024*1023/2
#define SLOPE   0.01f
#define EPS_BN  1e-5f
#define Y2_ST   36       // Y2 LDS row stride (16B aligned, conflict-free mod 64 banks)

// slope < 1  =>  leaky(v) == max(v, slope*v); lowers to v_mul + v_max_num_f32
__device__ __forceinline__ float leaky(float v) { return fmaxf(v, SLOPE * v); }

// ---------------------------------------------------------------- K1: BN(feat) -> x, node[:, :36]; cells = feat[:, :5]
__global__ __launch_bounds__(256) void k_bn(const float* __restrict__ feat,
                                            const float* __restrict__ g,
                                            const float* __restrict__ b,
                                            float* __restrict__ x,
                                            float* __restrict__ node,
                                            float* __restrict__ cells) {
  __shared__ float rs[256], rq[256];
  __shared__ float sc_sh, bi_sh;
  const int c = blockIdx.x, t = threadIdx.x;
  float s = 0.f, q = 0.f;
  for (int r = t; r < NN; r += 256) { float f = feat[r * D_IN + c]; s += f; q += f * f; }
  rs[t] = s; rq[t] = q; __syncthreads();
  for (int o = 128; o > 0; o >>= 1) {
    if (t < o) { rs[t] += rs[t + o]; rq[t] += rq[t + o]; }
    __syncthreads();
  }
  if (t == 0) {
    float m = rs[0] * (1.0f / NN);
    float v = rq[0] * (1.0f / NN) - m * m;
    float sc = rsqrtf(v + EPS_BN) * g[c];
    sc_sh = sc; bi_sh = b[c] - m * sc;
  }
  __syncthreads();
  const float sc = sc_sh, bi = bi_sh;
  for (int r = t; r < NN; r += 256) {
    float f = feat[r * D_IN + c];
    float xv = f * sc + bi;
    x[r * D_IN + c] = xv;
    node[r * NODE_ST + c] = xv;
    if (c < 5) cells[r * 5 + c] = f;
  }
}

// ---------------------------------------------------------------- K2: kNN (top-8 smallest L1 distance on first 4 dims)
__global__ __launch_bounds__(256) void k_knn(const float* __restrict__ x, int* __restrict__ idx) {
  __shared__ float x4[NN * 4];
  const int t = threadIdx.x;
  for (int e = t; e < NN * 4; e += 256) { int r = e >> 2, c = e & 3; x4[e] = x[r * D_IN + c]; }
  __syncthreads();
  const int i = blockIdx.x * 256 + t;
  const float xi0 = x4[i * 4 + 0], xi1 = x4[i * 4 + 1], xi2 = x4[i * 4 + 2], xi3 = x4[i * 4 + 3];
  float bd[KNN]; int bi[KNN];
#pragma unroll
  for (int s = 0; s < KNN; ++s) { bd[s] = 3.4e38f; bi[s] = 0; }
  for (int j = 0; j < NN; ++j) {
    float d = fabsf(xi0 - x4[j * 4 + 0]) + fabsf(xi1 - x4[j * 4 + 1]) +
              fabsf(xi2 - x4[j * 4 + 2]) + fabsf(xi3 - x4[j * 4 + 3]);
    if (d < bd[KNN - 1]) {
      float cd = d; int ci = j;
#pragma unroll
      for (int s = 0; s < KNN; ++s) {   // stable sorted insert (strict <) == top_k tie-break
        bool sw = cd < bd[s];
        float td = bd[s]; int ti = bi[s];
        if (sw) { bd[s] = cd; bi[s] = ci; cd = td; ci = ti; }
      }
    }
  }
#pragma unroll
  for (int s = 0; s < KNN; ++s) idx[i * KNN + s] = bi[s];
}

// ---------------------------------------------------------------- K3: edge-conv 1 (one wave per node)
__global__ __launch_bounds__(32) void k_edge1(const float* __restrict__ x, const int* __restrict__ idx,
                                              const float* __restrict__ w1, const float* __restrict__ b1,
                                              const float* __restrict__ w2, const float* __restrict__ b2,
                                              float* __restrict__ e1, float* __restrict__ node) {
  __shared__ float v[D_IN * KNN];   // 288, element c*8+k = x[nb[k], c]
  __shared__ float h[D_EDGE];
  __shared__ int nb[KNN];
  const int i = blockIdx.x, l = threadIdx.x;
  if (l < KNN) nb[l] = idx[i * KNN + l];
  __syncthreads();
  for (int t = l; t < D_IN * KNN; t += 32) { int c = t >> 3, k = t & 7; v[t] = x[nb[k] * D_IN + c]; }
  __syncthreads();
  float acc = b1[l];
  for (int t = 0; t < D_IN * KNN; ++t) acc += w1[l * (D_IN * KNN) + t] * v[t];
  h[l] = leaky(acc);
  __syncthreads();
  float acc2 = b2[l];
#pragma unroll 8
  for (int t = 0; t < D_EDGE; ++t) acc2 += w2[l * D_EDGE + t] * h[t];
  acc2 = leaky(acc2);
  e1[i * D_EDGE + l] = acc2;
  node[i * NODE_ST + D_IN + l] = acc2;
}

// ---------------------------------------------------------------- K4: edge-conv 2
__global__ __launch_bounds__(32) void k_edge2(const float* __restrict__ e1, const int* __restrict__ idx,
                                              const float* __restrict__ w1, const float* __restrict__ b1,
                                              const float* __restrict__ w2, const float* __restrict__ b2,
                                              float* __restrict__ node) {
  __shared__ float v[D_EDGE * KNN];   // 256
  __shared__ float h[D_EDGE];
  __shared__ int nb[KNN];
  const int i = blockIdx.x, l = threadIdx.x;
  if (l < KNN) nb[l] = idx[i * KNN + l];
  __syncthreads();
  for (int t = l; t < D_EDGE * KNN; t += 32) { int c = t >> 3, k = t & 7; v[t] = e1[nb[k] * D_EDGE + c]; }
  __syncthreads();
  float acc = b1[l];
  for (int t = 0; t < D_EDGE * KNN; ++t) acc += w1[l * (D_EDGE * KNN) + t] * v[t];
  h[l] = leaky(acc);
  __syncthreads();
  float acc2 = b2[l];
#pragma unroll 8
  for (int t = 0; t < D_EDGE; ++t) acc2 += w2[l * D_EDGE + t] * h[t];
  node[i * NODE_ST + D_IN + D_EDGE + l] = leaky(acc2);
}

// ---------------------------------------------------------------- K5: pair-BN stats as weighted node sums -> scale/bias (200)
__global__ __launch_bounds__(256) void k_stats(const float* __restrict__ node,
                                               const float* __restrict__ lg, const float* __restrict__ lb,
                                               float* __restrict__ scale, float* __restrict__ bias) {
  __shared__ float4 red[256];
  const int f = blockIdx.x, t = threadIdx.x;
  float s1 = 0.f, q1 = 0.f, s2 = 0.f, q2 = 0.f;
  for (int i = t; i < NN; i += 256) {
    float v = node[i * NODE_ST + f];
    float wa = (float)(NN - 1 - i), wb = (float)i;   // ti multiplicity, tj multiplicity
    s1 += wa * v; q1 += wa * v * v; s2 += wb * v; q2 += wb * v * v;
  }
  red[t] = make_float4(s1, q1, s2, q2); __syncthreads();
  for (int o = 128; o > 0; o >>= 1) {
    if (t < o) {
      float4 a = red[t], b = red[t + o];
      red[t] = make_float4(a.x + b.x, a.y + b.y, a.z + b.z, a.w + b.w);
    }
    __syncthreads();
  }
  if (t == 0) {
    const float invM = 1.0f / (float)M_PAIRS;
    float4 r = red[0];
    float m1 = r.x * invM, v1 = r.y * invM - m1 * m1;
    float sc1 = lg[f] * rsqrtf(v1 + EPS_BN);
    scale[f] = sc1; bias[f] = lb[f] - m1 * sc1;
    float m2 = r.z * invM, v2 = r.w * invM - m2 * m2;
    float sc2 = lg[100 + f] * rsqrtf(v2 + EPS_BN);
    scale[100 + f] = sc2; bias[100 + f] = lb[100 + f] - m2 * sc2;
  }
}

// ---------------------------------------------------------------- K6: folded bias  b1' = l_b1 + W1 @ bias
__global__ __launch_bounds__(32) void k_b1p(const float* __restrict__ w1, const float* __restrict__ b1,
                                            const float* __restrict__ bias, float* __restrict__ b1p) {
  const int o = threadIdx.x;
  float acc = b1[o];
  for (int f = 0; f < 200; ++f) acc += w1[o * 200 + f] * bias[f];
  b1p[o] = acc;
}

// ---------------------------------------------------------------- K7: per-node A = node@(W1a*scale)^T, B = node@(W1b*scale)^T
__global__ __launch_bounds__(256) void k_ab(const float* __restrict__ node, const float* __restrict__ w1,
                                            const float* __restrict__ scale,
                                            float* __restrict__ Aarr, float* __restrict__ Barr) {
  const int i = blockIdx.x * 8 + (threadIdx.x >> 5);
  const int o = threadIdx.x & 31;
  float accA = 0.f, accB = 0.f;
  for (int f = 0; f < 100; ++f) {
    float nf = node[i * NODE_ST + f];
    accA += nf * w1[o * 200 + f] * scale[f];
    accB += nf * w1[o * 200 + 100 + f] * scale[100 + f];
  }
  Aarr[i * 32 + o] = accA;
  Barr[i * 32 + o] = accB;
}

// ---------------------------------------------------------------- K8: pair kernel — WMMA f16 16x16x32 for the 32x32 layer
// 1023 blocks x 8 waves x 4 tiles x 16 pairs = 523,776 pairs (exact).
__global__ __launch_bounds__(256) void k_pairs(const float* __restrict__ Aarr, const float* __restrict__ Barr,
                                               const float* __restrict__ b1p, const float* __restrict__ w2,
                                               const float* __restrict__ b2, const float* __restrict__ w3,
                                               const float* __restrict__ b3, float* __restrict__ out) {
  __shared__ float y2s[8][16][Y2_ST];   // stride 36: 16B-aligned rows, 36n mod 64 distinct for n=0..15
  const int tid  = threadIdx.x;
  const int wv   = tid >> 5;
  const int lane = tid & 31;
  const int half = lane >> 4;        // 0 or 1
  const int n    = lane & 15;        // output col (B operand) / pair row (A operand)
  const int kbB  = half * 16;        // unused: B operand uses the kbA K chunks, same as A
  const int kbA  = half * 8;         // A operand first K chunk offset (second = +16)

  // B operands: cols n (WMMA#1) and n+16 (WMMA#2) of W2^T  ->  B[k][col] = W2[col][k]
  v16h bw1, bw2;
#pragma unroll
  for (int t = 0; t < 16; ++t) {
    bw1[t] = (_Float16)w2[n * 32 + kbA + t + (t >= 8 ? 8 : 0)];
    bw2[t] = (_Float16)w2[(n + 16) * 32 + kbA + t + (t >= 8 ? 8 : 0)];
  }
  // bias of layer 2 injected via the C operand (constant across rows of a lane's column)
  v8f cb1, cb2;
  {
    float bb1 = b2[n], bb2 = b2[n + 16];
#pragma unroll
    for (int r = 0; r < 8; ++r) { cb1[r] = bb1; cb2[r] = bb2; }
  }
  // layer-3 row for this lane's class (lanes 0-15 -> class 0, 16-31 -> class 1)
  float w3r[32];
#pragma unroll
  for (int k = 0; k < 32; ++k) w3r[k] = w3[half * 32 + k];
  const float b3c = b3[half];
  // folded layer-1 bias slices matching the A-operand K chunks
  float hb[16];
#pragma unroll
  for (int u = 0; u < 8; ++u) { hb[u] = b1p[kbA + u]; hb[8 + u] = b1p[kbA + 16 + u]; }

  const int tile0 = (blockIdx.x * 8 + wv) * 4;
  for (int tt = 0; tt < 4; ++tt) {
    const int p0 = (tile0 + tt) * 16;
    const int p  = p0 + n;                       // this lane's pair
    // inverse triangular map p -> (i, j), triu k=1, row-major
    float disc = 2047.0f * 2047.0f - 8.0f * (float)p;
    int i = (int)((2047.0f - sqrtf(disc)) * 0.5f);
    i = i < 0 ? 0 : (i > 1022 ? 1022 : i);
    while (((i + 1) * (2047 - (i + 1))) / 2 <= p) ++i;
    while ((i * (2047 - i)) / 2 > p) --i;
    const int j = i + 1 + (p - (i * (2047 - i)) / 2);

    // y1 = leaky(A[i] + B[j] + b1')  — build A operand (16x32 f16 layout)
    const float4* Ai = (const float4*)(Aarr + i * 32);
    const float4* Bj = (const float4*)(Barr + j * 32);
    const int q = kbA >> 2;                      // 0 or 2
    float4 a0 = Ai[q], a1 = Ai[q + 1], a2 = Ai[q + 4], a3 = Ai[q + 5];
    float4 c0 = Bj[q], c1 = Bj[q + 1], c2 = Bj[q + 4], c3 = Bj[q + 5];
    float yv[16];
    yv[0] = a0.x + c0.x; yv[1] = a0.y + c0.y; yv[2]  = a0.z + c0.z; yv[3]  = a0.w + c0.w;
    yv[4] = a1.x + c1.x; yv[5] = a1.y + c1.y; yv[6]  = a1.z + c1.z; yv[7]  = a1.w + c1.w;
    yv[8] = a2.x + c2.x; yv[9] = a2.y + c2.y; yv[10] = a2.z + c2.z; yv[11] = a2.w + c2.w;
    yv[12] = a3.x + c3.x; yv[13] = a3.y + c3.y; yv[14] = a3.z + c3.z; yv[15] = a3.w + c3.w;
    v16h av;
#pragma unroll
    for (int u = 0; u < 16; ++u) av[u] = (_Float16)leaky(yv[u] + hb[u]);

    // layer 2: Y2(16x32) = leaky(Y1 @ W2^T + b2) via two WMMAs
    v8f d1 = __builtin_amdgcn_wmma_f32_16x16x32_f16(false, av, false, bw1, (short)0, cb1, false, false);
    v8f d2 = __builtin_amdgcn_wmma_f32_16x16x32_f16(false, av, false, bw2, (short)0, cb2, false, false);
#pragma unroll
    for (int r = 0; r < 8; ++r) {
      const int rowM = r + half * 8;             // C/D layout: lanes<16 rows 0-7, lanes>=16 rows 8-15
      y2s[wv][rowM][n]      = leaky(d1[r]);
      y2s[wv][rowM][n + 16] = leaky(d2[r]);
    }
    __syncthreads();
    // layer 3: out[p, cls] = b3 + Y2[row] . W3[cls]  (b128 LDS reads)
    const float4* yrow = (const float4*)(&y2s[wv][n][0]);
    float acc = b3c;
#pragma unroll
    for (int kq = 0; kq < 8; ++kq) {
      float4 yq = yrow[kq];
      acc += yq.x * w3r[kq * 4 + 0] + yq.y * w3r[kq * 4 + 1] +
             yq.z * w3r[kq * 4 + 2] + yq.w * w3r[kq * 4 + 3];
    }
    out[(p0 + n) * 2 + half] = acc;
    __syncthreads();
  }
}

// ---------------------------------------------------------------- host
extern "C" void kernel_launch(void* const* d_in, const int* in_sizes, int n_in,
                              void* d_out, int out_size, void* d_ws, size_t ws_size,
                              hipStream_t stream) {
  const float* feat   = (const float*)d_in[0];
  const float* bn_g   = (const float*)d_in[1];
  const float* bn_b   = (const float*)d_in[2];
  const float* ec1_w1 = (const float*)d_in[3];
  const float* ec1_b1 = (const float*)d_in[4];
  const float* ec1_w2 = (const float*)d_in[5];
  const float* ec1_b2 = (const float*)d_in[6];
  const float* ec2_w1 = (const float*)d_in[7];
  const float* ec2_b1 = (const float*)d_in[8];
  const float* ec2_w2 = (const float*)d_in[9];
  const float* ec2_b2 = (const float*)d_in[10];
  const float* lbn_g  = (const float*)d_in[11];
  const float* lbn_b  = (const float*)d_in[12];
  const float* l_w1   = (const float*)d_in[13];
  const float* l_b1   = (const float*)d_in[14];
  const float* l_w2   = (const float*)d_in[15];
  const float* l_b2   = (const float*)d_in[16];
  const float* l_w3   = (const float*)d_in[17];
  const float* l_b3   = (const float*)d_in[18];

  float* out   = (float*)d_out;
  float* cells = out + (size_t)M_PAIRS * 2;

  float* ws   = (float*)d_ws;
  float* x    = ws;                       // 1024*36
  float* node = x + 1024 * 36;            // 1024*112
  float* e1   = node + 1024 * NODE_ST;    // 1024*32
  float* Aarr = e1 + 1024 * 32;           // 1024*32 (16B aligned)
  float* Barr = Aarr + 1024 * 32;         // 1024*32
  float* scl  = Barr + 1024 * 32;         // 200
  float* bias = scl + 200;                // 200
  float* b1p  = bias + 200;               // 32
  int*   idx  = (int*)(b1p + 32);         // 1024*8

  k_bn   <<<36,   256, 0, stream>>>(feat, bn_g, bn_b, x, node, cells);
  k_knn  <<<4,    256, 0, stream>>>(x, idx);
  k_edge1<<<1024, 32,  0, stream>>>(x, idx, ec1_w1, ec1_b1, ec1_w2, ec1_b2, e1, node);
  k_edge2<<<1024, 32,  0, stream>>>(e1, idx, ec2_w1, ec2_b1, ec2_w2, ec2_b2, node);
  k_stats<<<100,  256, 0, stream>>>(node, lbn_g, lbn_b, scl, bias);
  k_b1p  <<<1,    32,  0, stream>>>(l_w1, l_b1, bias, b1p);
  k_ab   <<<128,  256, 0, stream>>>(node, l_w1, scl, Aarr, Barr);
  k_pairs<<<1023, 256, 0, stream>>>(Aarr, Barr, b1p, l_w2, l_b2, l_w3, l_b3, out);
}
